// CapsuleNetwork_42700564857425
// MI455X (gfx1250) — matrix operand order repaired and checked by reading, not compiled, not run on hardware
//
#include <hip/hip_runtime.h>
#include <hip/hip_bf16.h>
#include <math.h>

typedef __attribute__((ext_vector_type(16))) __bf16 v16bf;
typedef __attribute__((ext_vector_type(8)))  __bf16 v8bf;
typedef __attribute__((ext_vector_type(8)))  float  v8f;

#define BN   64      // batch
#define SN   512     // sequence
#define DN   256     // c_in
#define COUT 256     // c_out
#define LOUT 256     // conv output length
#define PN   8192    // primary capsules (32 types * 256)
#define ON   5       // output capsules
#define ODIM 16
#define PDIM 8
#define EPSF 1e-8f
#define NT   4       // N sub-tiles per wave (16x64 strip)

// ---------------------------------------------------------------------------
// 1) f32 -> bf16 copy of x_input [B,S,D] (D = c_in contiguous: already the
//    B-operand-friendly layout for the implicit GEMM).
__global__ __launch_bounds__(256) void convert_x_kernel(
    const float* __restrict__ x, __bf16* __restrict__ xbf) {
  int i = blockIdx.x * 256 + threadIdx.x;
  if (i < BN * SN * DN) xbf[i] = (__bf16)x[i];
}

// 2) Repack conv_w [c_out][c_in][9] f32 -> wpack [k][c_out][c_in] bf16
__global__ __launch_bounds__(256) void pack_w_kernel(
    const float* __restrict__ w, __bf16* __restrict__ wp) {
  int i = blockIdx.x * 256 + threadIdx.x;
  if (i >= 9 * COUT * DN) return;
  int k   = i / (COUT * DN);
  int rem = i % (COUT * DN);
  int co  = rem / DN;
  int ci  = rem % DN;
  wp[i] = (__bf16)w[(co * DN + ci) * 9 + k];
}

// ---------------------------------------------------------------------------
// 3) Conv1d as implicit GEMM via v_wmma_f32_16x16x32_bf16.
//    M = c_out (256), N = b*256+l (16384), K = c_in (256) x 9 taps.
//    Each wave computes a 16x64 strip (4 accumulators): the A fragment is
//    loaded once per K-step and reused by 4 back-to-back WMMAs. Epilogue adds
//    bias and scatters directly into capsule layout pcbuf[b][p][pd].
__global__ __launch_bounds__(256) void conv_wmma_kernel(
    const __bf16* __restrict__ xbf, const __bf16* __restrict__ wp,
    const float* __restrict__ bias, float* __restrict__ pcbuf) {
  const int wave = threadIdx.x >> 5;
  const int lane = threadIdx.x & 31;
  const int tile = blockIdx.x * 8 + wave;   // 0..4095
  const int m0   = (tile & 15) * 16;        // c_out tile base
  const int n0   = (tile >> 4) * (16 * NT); // (b,l) strip base (64-wide)
  const int l16  = lane & 15;
  const int hi   = lane >> 4;               // 0 or 1

  // Per-sub-tile coordinates; a 64-wide strip never crosses a batch boundary.
  int bimg[NT], li[NT];
#pragma unroll
  for (int t = 0; t < NT; ++t) {
    int n   = n0 + t * 16 + l16;
    bimg[t] = n >> 8;
    li[t]   = n & 255;
  }

  v8f acc[NT];
#pragma unroll
  for (int t = 0; t < NT; ++t) acc[t] = (v8f)0.0f;

  // A row for this lane: M = l16; hi selects the K-half offset (+16 halves)
  const __bf16* arow = wp + (m0 + l16) * DN + hi * 16;

  for (int k = 0; k < 9; ++k) {
    const __bf16* ak = arow + k * (COUT * DN);
    const __bf16* brow[NT];
    bool          valid[NT];
#pragma unroll
    for (int t = 0; t < NT; ++t) {
      int pos  = 2 * li[t] + k - 4;
      valid[t] = (pos >= 0) && (pos < SN);
      brow[t]  = xbf + ((size_t)bimg[t] * SN + pos) * DN + hi * 16;
    }
#pragma unroll
    for (int kk0 = 0; kk0 < DN; kk0 += 32) {
      // A fragment: lane<16 holds K = kk0+{0..15}; lane>=16 K+16.
      v8bf alo = *(const v8bf*)(ak + kk0);
      v8bf ahi = *(const v8bf*)(ak + kk0 + 8);
      v16bf a;
#pragma unroll
      for (int i = 0; i < 8; ++i) { a[i] = alo[i]; a[i + 8] = ahi[i]; }
      // B fragments: lane l: N = l%16, K = kk0 + (l<16?0:16) + j (contiguous)
#pragma unroll
      for (int t = 0; t < NT; ++t) {
        v16bf bv = (v16bf)(__bf16)0.0f;
        if (valid[t]) bv = *(const v16bf*)(brow[t] + kk0);
        acc[t] = __builtin_amdgcn_wmma_f32_16x16x32_bf16(
            false, a, false, bv, (short)0, acc[t], false, false);
      }
    }
  }
  // D layout: lane l, reg r -> M = r + 8*(l>=16), N = l%16
#pragma unroll
  for (int r = 0; r < 8; ++r) {
    int   co   = m0 + r + hi * 8;
    float bco  = bias[co];
    int   type = co >> 3, pd = co & 7;
#pragma unroll
    for (int t = 0; t < NT; ++t) {
      int p = type * 256 + li[t];
      pcbuf[((size_t)bimg[t] * PN + p) * PDIM + pd] = acc[t][r] + bco;
    }
  }
}

// ---------------------------------------------------------------------------
__device__ __forceinline__ float squash_scale(float sq) {
  return (sq / (1.0f + sq)) / (sqrtf(sq + EPSF) + EPSF);
}

// 4) Squash primary capsules: u_i = squash(pc) over the 8-dim vector
__global__ __launch_bounds__(256) void squash_u_kernel(
    const float* __restrict__ pc, float* __restrict__ ui) {
  int t = blockIdx.x * 256 + threadIdx.x;   // b*8192+p
  if (t >= BN * PN) return;
  const float* src = pc + (size_t)t * PDIM;
  float v[PDIM], sq = 0.0f;
#pragma unroll
  for (int e = 0; e < PDIM; ++e) { v[e] = src[e]; sq += v[e] * v[e]; }
  float s = squash_scale(sq);
  float* dst = ui + (size_t)t * PDIM;
#pragma unroll
  for (int e = 0; e < PDIM; ++e) dst[e] = v[e] * s;
}

// 5) u_hat[b,p,o,d] = sum_e W[p,o,d,e] * u_i[b,p,e]
__global__ __launch_bounds__(256) void uhat_kernel(
    const float* __restrict__ ui, const float* __restrict__ W,
    float* __restrict__ uhat) {
  int t = blockIdx.x * 256 + threadIdx.x;   // (b*8192+p)*5+o
  if (t >= BN * PN * ON) return;
  int o  = t % ON;
  int bp = t / ON;
  int p  = bp & (PN - 1);
  float ue[PDIM];
  const float* u = ui + (size_t)bp * PDIM;
#pragma unroll
  for (int e = 0; e < PDIM; ++e) ue[e] = u[e];
  const float* Wp  = W + ((size_t)(p * ON + o) * ODIM) * PDIM;
  float*       out = uhat + (size_t)bp * (ON * ODIM) + o * ODIM;
#pragma unroll
  for (int d = 0; d < ODIM; ++d) {
    float s = 0.0f;
#pragma unroll
    for (int e = 0; e < PDIM; ++e) s += Wp[d * PDIM + e] * ue[e];
    out[d] = s;
  }
}

// 6) s_j[b,o,:] = sum_p c[b,p,o] * u_hat[b,p,o,:]   (c = 0.2 when use_c==0)
__global__ __launch_bounds__(256) void sj_kernel(
    const float* __restrict__ uhat, const float* __restrict__ cbuf,
    float* __restrict__ sbuf, int use_c) {
  __shared__ float red[256 * ODIM];
  int b = blockIdx.x / ON, o = blockIdx.x % ON;
  float acc[ODIM];
#pragma unroll
  for (int d = 0; d < ODIM; ++d) acc[d] = 0.0f;
  for (int p = threadIdx.x; p < PN; p += 256) {
    size_t bp = (size_t)b * PN + p;
    float  c  = use_c ? cbuf[bp * ON + o] : 0.2f;
    const float* uh = uhat + bp * (ON * ODIM) + o * ODIM;
#pragma unroll
    for (int d = 0; d < ODIM; ++d) acc[d] += c * uh[d];
  }
#pragma unroll
  for (int d = 0; d < ODIM; ++d) red[threadIdx.x * ODIM + d] = acc[d];
  __syncthreads();
  for (int off = 128; off > 0; off >>= 1) {
    if ((int)threadIdx.x < off)
#pragma unroll
      for (int d = 0; d < ODIM; ++d)
        red[threadIdx.x * ODIM + d] += red[(threadIdx.x + off) * ODIM + d];
    __syncthreads();
  }
  if (threadIdx.x < ODIM)
    sbuf[(b * ON + o) * ODIM + threadIdx.x] = red[threadIdx.x];
}

// 7) v_j = squash(s_j) over the 16-dim vector (also used for final output)
__global__ __launch_bounds__(64) void squash_v_kernel(
    const float* __restrict__ sbuf, float* __restrict__ vbuf) {
  int t = blockIdx.x * 64 + threadIdx.x;    // b*5+o
  if (t >= BN * ON) return;
  const float* s = sbuf + t * ODIM;
  float v[ODIM], sq = 0.0f;
#pragma unroll
  for (int d = 0; d < ODIM; ++d) { v[d] = s[d]; sq += v[d] * v[d]; }
  float sc = squash_scale(sq);
  float* dst = vbuf + t * ODIM;
#pragma unroll
  for (int d = 0; d < ODIM; ++d) dst[d] = v[d] * sc;
}

// 8) b_ij update + softmax over the 5 output capsules
__global__ __launch_bounds__(256) void logits_kernel(
    const float* __restrict__ uhat, const float* __restrict__ vbuf,
    float* __restrict__ bij, float* __restrict__ cbuf, int first) {
  int t = blockIdx.x * 256 + threadIdx.x;   // b*8192+p
  if (t >= BN * PN) return;
  int b = t >> 13;
  const float* uh = uhat + (size_t)t * (ON * ODIM);
  const float* v  = vbuf + b * (ON * ODIM);
  float bo[ON];
#pragma unroll
  for (int o = 0; o < ON; ++o) {
    float dot = 0.0f;
#pragma unroll
    for (int d = 0; d < ODIM; ++d) dot += uh[o * ODIM + d] * v[o * ODIM + d];
    bo[o] = (first ? 0.0f : bij[(size_t)t * ON + o]) + dot;
    bij[(size_t)t * ON + o] = bo[o];
  }
  float m = bo[0];
#pragma unroll
  for (int o = 1; o < ON; ++o) m = fmaxf(m, bo[o]);
  float e[ON], s = 0.0f;
#pragma unroll
  for (int o = 0; o < ON; ++o) { e[o] = expf(bo[o] - m); s += e[o]; }
  float inv = 1.0f / s;
#pragma unroll
  for (int o = 0; o < ON; ++o) cbuf[(size_t)t * ON + o] = e[o] * inv;
}

// ---------------------------------------------------------------------------
extern "C" void kernel_launch(void* const* d_in, const int* in_sizes, int n_in,
                              void* d_out, int out_size, void* d_ws, size_t ws_size,
                              hipStream_t stream) {
  const float* x_input = (const float*)d_in[0];   // [64,512,256]
  const float* conv_w  = (const float*)d_in[1];   // [256,256,9]
  const float* conv_b  = (const float*)d_in[2];   // [256]
  const float* Wmat    = (const float*)d_in[3];   // [8192,5,16,8]

  char*  ws  = (char*)d_ws;
  size_t off = 0;
  __bf16* xbf  = (__bf16*)(ws + off); off += (size_t)BN * SN * DN * 2;
  __bf16* wpk  = (__bf16*)(ws + off); off += (size_t)9 * COUT * DN * 2;
  float*  pcb  = (float*)(ws + off);  off += (size_t)BN * PN * PDIM * 4;
  float*  uib  = (float*)(ws + off);  off += (size_t)BN * PN * PDIM * 4;
  float*  uhat = (float*)(ws + off);  off += (size_t)BN * PN * ON * ODIM * 4;
  float*  bij  = (float*)(ws + off);  off += (size_t)BN * PN * ON * 4;
  float*  cb   = (float*)(ws + off);  off += (size_t)BN * PN * ON * 4;
  float*  sb   = (float*)(ws + off);  off += (size_t)BN * ON * ODIM * 4;
  float*  vb   = (float*)(ws + off);  off += (size_t)BN * ON * ODIM * 4;

  convert_x_kernel<<<(BN * SN * DN) / 256, 256, 0, stream>>>(x_input, xbf);
  pack_w_kernel<<<(9 * COUT * DN) / 256, 256, 0, stream>>>(conv_w, wpk);
  // 16 M-tiles * 256 N-strips = 4096 wave-tiles, 8 waves/block -> 512 blocks
  conv_wmma_kernel<<<512, 256, 0, stream>>>(xbf, wpk, conv_b, pcb);
  squash_u_kernel<<<(BN * PN) / 256, 256, 0, stream>>>(pcb, uib);
  uhat_kernel<<<(BN * PN * ON) / 256, 256, 0, stream>>>(uib, Wmat, uhat);

  // routing iter 0: uniform c = 0.2
  sj_kernel<<<BN * ON, 256, 0, stream>>>(uhat, cb, sb, 0);
  squash_v_kernel<<<(BN * ON + 63) / 64, 64, 0, stream>>>(sb, vb);
  // iter 1
  logits_kernel<<<(BN * PN) / 256, 256, 0, stream>>>(uhat, vb, bij, cb, 1);
  sj_kernel<<<BN * ON, 256, 0, stream>>>(uhat, cb, sb, 1);
  squash_v_kernel<<<(BN * ON + 63) / 64, 64, 0, stream>>>(sb, vb);
  // iter 2 (final v written straight to d_out)
  logits_kernel<<<(BN * PN) / 256, 256, 0, stream>>>(uhat, vb, bij, cb, 0);
  sj_kernel<<<BN * ON, 256, 0, stream>>>(uhat, cb, sb, 1);
  squash_v_kernel<<<(BN * ON + 63) / 64, 64, 0, stream>>>(sb, (float*)d_out);
}
